// ParallelSlowCastle_23519240913261
// MI455X (gfx1250) — matrix operand order repaired and checked by reading, not compiled, not run on hardware
//
#include <hip/hip_runtime.h>
#include <hip/hip_bf16.h>

// ---------------------------------------------------------------------------
// ParallelSlowCastle attention for MI455X (gfx1250, wave32, WMMA bf16).
//
// v3: k_flash widened to 16x64 S_u blocks (4 k-tiles). The triangular masks
// are baked into the materialized term1/sig matrices, so the j-contraction is
// one clean GEMM loop (A-frag reused 4x, no boundary zeroing), softmax state
// updates 4x less often, and P@Vc runs at full K=64 (no zero padding).
// global_prefetch_b8 issued one chunk ahead on the hot A/B streams.
// ---------------------------------------------------------------------------

#define BATCH 2
#define SEQ   2048
#define DIM   512
#define HEADS 8
#define DH    64
#define BH    (BATCH * HEADS)
#define SCALE 0.125f
#define NEG_INF (-__builtin_inff())

typedef __attribute__((ext_vector_type(8)))  __bf16 v8bf;
typedef __attribute__((ext_vector_type(16))) __bf16 v16bf;
typedef __attribute__((ext_vector_type(8)))  float  v8f;

// ws layout (bf16 elements):
//   [0, WT_ELEMS)            : 7 transposed bf16 weights (512x512 each)
//   qkv = +WT_ELEMS          : [0]=Qu [1]=Ku [2]=Vu [3]=Qc [4]=Kc
//                              [5]=VcT(b,h,d,n) [6]=attn_out   (7 * TEN)
//   band = qkv + 7*TEN       : per head-slot { term1[N][N], sig[N][N] } bf16
#define WT_ELEMS   (7 * 512 * 512)
#define TEN        (BH * SEQ * DH)            // 2,097,152
#define PER_HEAD   ((size_t)2 * SEQ * SEQ)    // term1 + sig, bf16 elems

__device__ __forceinline__ int lane_id() { return (int)(threadIdx.x & 31u); }

__device__ __forceinline__ v16bf make_frag(v8bf a, v8bf b) {
  v16bf r;
#pragma unroll
  for (int e = 0; e < 8; ++e) { r[e] = a[e]; r[e + 8] = b[e]; }
  return r;
}

// A fragment (16x32 bf16): row-major source, row = lane%16,
// K chunks {ko..ko+7} and {ko+16..ko+23}, ko = 0 (lanes 0-15) / 8 (lanes 16-31).
__device__ __forceinline__ v16bf load_a(const __bf16* base, int ld, int kb) {
  int l = lane_id();
  const __bf16* p = base + (size_t)(l & 15) * ld + kb + ((l < 16) ? 0 : 8);
  return make_frag(*(const v8bf*)p, *(const v8bf*)(p + 16));
}

// Same, sourcing fp32 and converting to bf16 on the fly.
__device__ __forceinline__ v16bf load_a_f32(const float* base, int ld, int kb) {
  int l = lane_id();
  const float* p = base + (size_t)(l & 15) * ld + kb + ((l < 16) ? 0 : 8);
  v16bf r;
#pragma unroll
  for (int e = 0; e < 8; ++e) { r[e] = (__bf16)p[e]; r[e + 8] = (__bf16)p[e + 16]; }
  return r;
}

// B fragment (32x16 bf16) from a TRANSPOSED operand Bt[n][k] (row-major,
// stride ld): col = lane%16, K chunks {ko..ko+7} and {ko+16..ko+23}, ko = 0 (lanes 0-15) / 8 (lanes 16-31).
__device__ __forceinline__ v16bf load_bt(const __bf16* bt, int ld, int kb) {
  int l = lane_id();
  const __bf16* p = bt + (size_t)(l & 15) * ld + kb + ((l < 16) ? 0 : 8);
  return make_frag(*(const v8bf*)p, *(const v8bf*)(p + 16));
}

__device__ __forceinline__ v8f wmma_bf16(v16bf a, v16bf b, v8f c) {
  return __builtin_amdgcn_wmma_f32_16x16x32_bf16(false, a, false, b, (short)0, c,
                                                 false, false);
}

// ---------------------------------------------------------------------------
// Kernel 1: transpose + convert the 7 weight matrices to bf16.
// wt[w][n][k] = W[w][k][n]
// ---------------------------------------------------------------------------
__global__ __launch_bounds__(256) void k_prep_weights(
    const float* w0, const float* w1, const float* w2, const float* w3,
    const float* w4, const float* w5, const float* w6, __bf16* wt) {
  int idx = blockIdx.x * 256 + threadIdx.x;          // 7*512*512 total
  const float* ws[7] = {w0, w1, w2, w3, w4, w5, w6};
  int w = idx >> 18;
  int rem = idx & 262143;
  int n = rem >> 9;
  int k = rem & 511;
  wt[idx] = (__bf16)ws[w][k * 512 + n];
}

// ---------------------------------------------------------------------------
// Kernel 2: six input projections.  One wave = 16x64 output strip.
// Vc (widx 5) is written transposed per head: VcT[b][h][d][n].
// grid = (32, 8, 6), block = 256 (8 waves).
// ---------------------------------------------------------------------------
__global__ __launch_bounds__(256) void k_proj(const float* __restrict__ x,
                                              const __bf16* __restrict__ wt,
                                              __bf16* __restrict__ qkv) {
  int wave = threadIdx.x >> 5;
  int mtile = blockIdx.x * 8 + wave;                 // 256 m-tiles (M = 4096)
  int nbase = blockIdx.y * 64;
  int widx = blockIdx.z;
  const __bf16* wtw = wt + (size_t)widx * 512 * 512;
  const float* xa = x + (size_t)mtile * 16 * 512;

  v8f acc[4] = {};
#pragma unroll 4
  for (int kb = 0; kb < 512; kb += 32) {
    v16bf a = load_a_f32(xa, 512, kb);
#pragma unroll
    for (int t = 0; t < 4; ++t) {
      v16bf b = load_bt(wtw + (size_t)(nbase + t * 16) * 512, 512, kb);
      acc[t] = wmma_bf16(a, b, acc[t]);
    }
  }

  int lane = lane_id();
  int colL = lane & 15;
  int rowoff = (lane >> 4) * 8;
  __bf16* dst = qkv + (size_t)widx * TEN;
#pragma unroll
  for (int t = 0; t < 4; ++t) {
#pragma unroll
    for (int r = 0; r < 8; ++r) {
      int m = mtile * 16 + r + rowoff;               // m = b*SEQ + nseq
      int b = m >> 11;
      int nseq = m & 2047;
      int c = nbase + t * 16 + colL;                 // c = h*DH + d
      int h = c >> 6;
      int d = c & 63;
      __bf16 v = (__bf16)acc[t][r];
      if (widx == 5)
        dst[((size_t)(b * HEADS + h) * DH + d) * SEQ + nseq] = v;     // VcT
      else
        dst[((size_t)(b * HEADS + h) * SEQ + nseq) * DH + d] = v;
    }
  }
}

// ---------------------------------------------------------------------------
// Kernel 3: build the masked term1 and sigmoid matrices (bf16, row-major).
//   mat 0: term1[i][j] = (j<=i) ? (Qc_i . Vu_j)*SCALE : 0
//   mat 1: sig  [k][j] = (j>k)  ? sigmoid((Qu_k . Ku_j)*SCALE) : 0
// One wave = 16x64 tile; tiles that k_flash never reads are skipped.
// grid = (16, 32, 2*nh), block = 256.
// ---------------------------------------------------------------------------
__global__ __launch_bounds__(256) void k_build(const __bf16* __restrict__ qkv,
                                               __bf16* __restrict__ band, int hb) {
  int wave = threadIdx.x >> 5;
  int it = blockIdx.x * 8 + wave;                    // row-tile (i or k), 0..127
  int jb64 = blockIdx.y * 64;
  int hs = blockIdx.z >> 1;
  int mat = blockIdx.z & 1;
  int h = hb + hs;

  if (mat == 0) {                                    // term1: band around/below diag
    if (jb64 > it * 16 + 31) return;
  } else {                                           // sig: at/above diag only
    if (jb64 + 63 < it * 16) return;
  }

  const __bf16* A =
      qkv + (size_t)(mat ? 0 : 3) * TEN + (size_t)h * SEQ * DH +
      (size_t)it * 16 * DH;                          // Qu (sig) or Qc (term1)
  const __bf16* Bsrc =
      qkv + (size_t)(mat ? 1 : 2) * TEN + (size_t)h * SEQ * DH;  // Ku or Vu

  v16bf a0 = load_a(A, DH, 0);
  v16bf a1 = load_a(A, DH, 32);
  v8f acc[4] = {};
#pragma unroll
  for (int t = 0; t < 4; ++t) {
    const __bf16* bj = Bsrc + (size_t)(jb64 + t * 16) * DH;
    acc[t] = wmma_bf16(a0, load_bt(bj, DH, 0), acc[t]);
    acc[t] = wmma_bf16(a1, load_bt(bj, DH, 32), acc[t]);
  }

  int lane = lane_id();
  int colL = lane & 15;
  int rowoff = (lane >> 4) * 8;
  __bf16* dst = band + (size_t)hs * PER_HEAD + (size_t)mat * SEQ * SEQ;
#pragma unroll
  for (int t = 0; t < 4; ++t) {
#pragma unroll
    for (int r = 0; r < 8; ++r) {
      int rg = it * 16 + r + rowoff;                 // i (term1) / k (sig)
      int j = jb64 + t * 16 + colL;
      float v = acc[t][r] * SCALE;
      float o = mat ? ((j > rg) ? 1.f / (1.f + __expf(-v)) : 0.f)
                    : ((j <= rg) ? v : 0.f);
      dst[(size_t)rg * SEQ + j] = (__bf16)o;
    }
  }
}

// ---------------------------------------------------------------------------
// Kernel 4: flash pass.  One wave owns one (head-slot, 16-row i-tile) and
// walks 64-column k-blocks (4 k-tiles):
//   S_u 16x64: one GEMM over j = k0..it*16 (stored zeros absorb both
//              triangular boundaries; A-frag reused across the 4 k-tiles),
//   S_c 16x64: 8 WMMAs (Qc . Kc^T),
//   online softmax of S_c - silu(S_u) over 64 cols,
//   P(16x64) @ Vc at full K=64 via LDS re-fragmentation (8 WMMAs).
// grid = nh*16 blocks x 256 threads.
// ---------------------------------------------------------------------------
__global__ __launch_bounds__(256) void k_flash(const __bf16* __restrict__ qkv,
                                               const __bf16* __restrict__ band,
                                               __bf16* __restrict__ attnout,
                                               int hb) {
  __shared__ __align__(16) __bf16 plds[8][16 * 64];  // 16 KB: P tile per wave

  int wave = threadIdx.x >> 5;
  int wid = blockIdx.x * 8 + wave;
  int it = wid & 127;
  int hs = wid >> 7;
  int h = hb + hs;

  const __bf16* Qc = qkv + 3 * (size_t)TEN + (size_t)h * SEQ * DH;
  const __bf16* Kc = qkv + 4 * (size_t)TEN + (size_t)h * SEQ * DH;
  const __bf16* VcT = qkv + 5 * (size_t)TEN + (size_t)h * DH * SEQ;
  const __bf16* T1 = band + (size_t)hs * PER_HEAD + (size_t)it * 16 * SEQ;
  const __bf16* SG = band + (size_t)hs * PER_HEAD + (size_t)SEQ * SEQ;

  __bf16* pl = &plds[wave][0];
  int lane = lane_id();
  int colL = lane & 15;
  int rowoff = (lane >> 4) * 8;

  v16bf qcA0 = load_a(Qc + (size_t)it * 16 * DH, DH, 0);
  v16bf qcA1 = load_a(Qc + (size_t)it * 16 * DH, DH, 32);

  float m_run[8], l_run[8];
  v8f acc[4] = {};
#pragma unroll
  for (int r = 0; r < 8; ++r) { m_run[r] = NEG_INF; l_run[r] = 0.f; }

  // per-lane base pointers for prefetch of the hot streams
  const __bf16* pfA = T1 + (size_t)(lane & 15) * SEQ;
  const __bf16* pfB = SG + (size_t)(lane & 15) * SEQ;

  for (int kb = 0; kb <= it; kb += 4) {              // 64-col k-block
    int k0 = kb * 16;

    // ---- S_u 16x64 block: term1[i-panel, j] @ sig^T, j = k0 .. it*16 (+31)
    v8f su[4] = {};
    for (int jb = k0; jb <= it * 16; jb += 32) {
      __builtin_prefetch(pfA + jb + 32, 0, 3);       // global_prefetch_b8
      __builtin_prefetch(pfB + (size_t)k0 * SEQ + jb + 32, 0, 3);
      v16bf tA = load_a(T1, SEQ, jb);
#pragma unroll
      for (int t = 0; t < 4; ++t) {
        v16bf sB = load_bt(SG + (size_t)(k0 + t * 16) * SEQ, SEQ, jb);
        su[t] = wmma_bf16(tA, sB, su[t]);
      }
    }

    // ---- S_c 16x64 block
    v8f sc[4];
#pragma unroll
    for (int t = 0; t < 4; ++t) {
      const __bf16* kc = Kc + (size_t)(kb + t) * 16 * DH;
      v8f c = {};
      c = wmma_bf16(qcA0, load_bt(kc, DH, 0), c);
      c = wmma_bf16(qcA1, load_bt(kc, DH, 32), c);
      sc[t] = c;
    }

    // ---- final scores (into sc), 64 cols
#pragma unroll
    for (int t = 0; t < 4; ++t) {
#pragma unroll
      for (int r = 0; r < 8; ++r) {
        int i_g = it * 16 + r + rowoff;
        int k_g = k0 + t * 16 + colL;
        float s_u = su[t][r];
        float silu = s_u / (1.f + __expf(-s_u));
        sc[t][r] = (k_g <= i_g) ? (sc[t][r] * SCALE - silu) : NEG_INF;
      }
    }

    // ---- online softmax over the 64-col block
#pragma unroll
    for (int r = 0; r < 8; ++r) {
      float rm = fmaxf(fmaxf(sc[0][r], sc[1][r]), fmaxf(sc[2][r], sc[3][r]));
      rm = fmaxf(rm, __shfl_xor(rm, 1));
      rm = fmaxf(rm, __shfl_xor(rm, 2));
      rm = fmaxf(rm, __shfl_xor(rm, 4));
      rm = fmaxf(rm, __shfl_xor(rm, 8));
      float m_new = fmaxf(m_run[r], rm);             // finite: k=0..i exists
      float scale = __expf(m_run[r] - m_new);
      int row = r + rowoff;
      float rs = 0.f;
#pragma unroll
      for (int t = 0; t < 4; ++t) {
        float p = __expf(sc[t][r] - m_new);
        rs += p;
        pl[row * 64 + t * 16 + colL] = (__bf16)p;
      }
      rs += __shfl_xor(rs, 1);
      rs += __shfl_xor(rs, 2);
      rs += __shfl_xor(rs, 4);
      rs += __shfl_xor(rs, 8);
      l_run[r] = l_run[r] * scale + rs;
      m_run[r] = m_new;
#pragma unroll
      for (int dt = 0; dt < 4; ++dt) acc[dt][r] *= scale;
    }

    // ---- P(16x64) @ Vc(64xDH) at full K
    v16bf pA0 = load_a(pl, 64, 0);
    v16bf pA1 = load_a(pl, 64, 32);
#pragma unroll
    for (int dt = 0; dt < 4; ++dt) {
      const __bf16* vb = VcT + (size_t)dt * 16 * SEQ;
      acc[dt] = wmma_bf16(pA0, load_bt(vb, SEQ, k0), acc[dt]);
      acc[dt] = wmma_bf16(pA1, load_bt(vb, SEQ, k0 + 32), acc[dt]);
    }
  }

  // normalize + store attention output (bf16, [b,h,n,d])
#pragma unroll
  for (int dt = 0; dt < 4; ++dt) {
#pragma unroll
    for (int r = 0; r < 8; ++r) {
      int row = r + rowoff;
      attnout[(size_t)h * SEQ * DH + (size_t)(it * 16 + row) * DH + dt * 16 + colL] =
          (__bf16)(acc[dt][r] / l_run[r]);
    }
  }
}

// ---------------------------------------------------------------------------
// Kernel 5: merge heads @ w_out^T -> fp32 output (B,N,512).
// grid = (32, 8), block = 256.
// ---------------------------------------------------------------------------
__global__ __launch_bounds__(256) void k_out(const __bf16* __restrict__ attnout,
                                             const __bf16* __restrict__ wt_out,
                                             float* __restrict__ out) {
  int wave = threadIdx.x >> 5;
  int mtile = blockIdx.x * 8 + wave;                 // 256 m-tiles
  int nbase = blockIdx.y * 64;
  int lane = lane_id();
  int colL = lane & 15;
  int rowoff = (lane >> 4) * 8;
  int m0 = mtile * 16;

  v8f acc[4] = {};
#pragma unroll 4
  for (int kb = 0; kb < 512; kb += 32) {
    // A fragment over k = h*64+d; each 8-half chunk stays inside one head.
    int row = lane & 15;
    int m = m0 + row;
    int b = m >> 11;
    int nseq = m & 2047;
    int k0 = kb + ((lane < 16) ? 0 : 8);
    int k1 = k0 + 16;
    const __bf16* p0 =
        attnout + ((size_t)(b * HEADS + (k0 >> 6)) * SEQ + nseq) * DH + (k0 & 63);
    const __bf16* p1 =
        attnout + ((size_t)(b * HEADS + (k1 >> 6)) * SEQ + nseq) * DH + (k1 & 63);
    v16bf a = make_frag(*(const v8bf*)p0, *(const v8bf*)p1);
#pragma unroll
    for (int t = 0; t < 4; ++t) {
      v16bf bfrag = load_bt(wt_out + (size_t)(nbase + t * 16) * 512, 512, kb);
      acc[t] = wmma_bf16(a, bfrag, acc[t]);
    }
  }
#pragma unroll
  for (int t = 0; t < 4; ++t) {
#pragma unroll
    for (int r = 0; r < 8; ++r) {
      out[(size_t)(m0 + r + rowoff) * 512 + nbase + t * 16 + colL] = acc[t][r];
    }
  }
}

// ---------------------------------------------------------------------------
extern "C" void kernel_launch(void* const* d_in, const int* in_sizes, int n_in,
                              void* d_out, int out_size, void* d_ws, size_t ws_size,
                              hipStream_t stream) {
  (void)in_sizes; (void)n_in; (void)out_size;
  const float* x = (const float*)d_in[0];
  const float* w_q_u = (const float*)d_in[1];
  const float* w_k_u = (const float*)d_in[2];
  const float* w_v_u = (const float*)d_in[3];
  const float* w_q_c = (const float*)d_in[4];
  const float* w_k_c = (const float*)d_in[5];
  const float* w_v_c = (const float*)d_in[6];
  const float* w_out = (const float*)d_in[7];
  float* out = (float*)d_out;

  __bf16* ws = (__bf16*)d_ws;
  __bf16* wt = ws;
  __bf16* qkv = ws + WT_ELEMS;
  __bf16* attnout = qkv + (size_t)6 * TEN;
  __bf16* band = qkv + (size_t)7 * TEN;

  // choose head-chunk size (16,8,4,2,1) that fits ws (+256 B tail slack for
  // the benign OOB fragment over-reads at the causal boundary).
  size_t base_bytes = (size_t)(WT_ELEMS + 7 * (size_t)TEN) * sizeof(__bf16);
  int nh = 16;
  while (nh > 1 &&
         base_bytes + (size_t)nh * PER_HEAD * sizeof(__bf16) + 256 > ws_size)
    nh >>= 1;

  k_prep_weights<<<7168, 256, 0, stream>>>(w_q_u, w_k_u, w_v_u, w_q_c, w_k_c,
                                           w_v_c, w_out, wt);
  dim3 g2(32, 8, 6);
  k_proj<<<g2, 256, 0, stream>>>(x, wt, qkv);

  for (int hb = 0; hb < BH; hb += nh) {
    dim3 gb(16, 32, 2 * nh);
    k_build<<<gb, 256, 0, stream>>>(qkv, band, hb);
    k_flash<<<nh * 16, 256, 0, stream>>>(qkv, band, attnout, hb);
  }

  dim3 g4(32, 8);
  k_out<<<g4, 256, 0, stream>>>(attnout, wt + (size_t)6 * 512 * 512, out);
}
